// SigmoidCrossAttention_36825049596578
// MI455X (gfx1250) — matrix operand order repaired and checked by reading, not compiled, not run on hardware
//
#include <hip/hip_runtime.h>
#include <hip/hip_bf16.h>
#include <math.h>

typedef __bf16 bf16_t;
typedef __attribute__((ext_vector_type(16))) __bf16 v16bf;
typedef __attribute__((ext_vector_type(8)))  float  v8f;

#define HEADS     16
#define HEAD_DIM  64
#define NQ        2048
#define NKK       2048
#define DMODEL    1024
#define AMODEL    1024
#define BATCH     2
#define MROWS     (BATCH * NQ)          // 4096
#define SIG_SCALE 0.35355339059327373f  // 64^-0.25
#define LN_EPS    1e-5f

// ---------------------------------------------------------------------------
// CDNA5 async memory->LDS copy (ASYNCcnt path). VDST operand carries the LDS
// byte address (generic shared ptr low 32 bits, per aperture mapping).
// ---------------------------------------------------------------------------
static __device__ __forceinline__ void async_copy_b128(unsigned ldsOff, const void* gp) {
  asm volatile("global_load_async_to_lds_b128 %0, %1, off"
               :: "v"(ldsOff), "v"(gp) : "memory");
}
static __device__ __forceinline__ void wait_async0() {
  asm volatile("s_wait_asynccnt 0x0" ::: "memory");
}

// ---------------------------------------------------------------------------
// helpers
// ---------------------------------------------------------------------------
static __device__ __forceinline__ v8f vzero8() {
  v8f z;
#pragma unroll
  for (int i = 0; i < 8; ++i) z[i] = 0.0f;
  return z;
}

// A-fragment (16x32 bf16, K-contiguous rows in LDS).
// ISA 7.12.2: lanes 0-15 hold row M=lane, K {0..7, 16..23}; lanes 16-31 hold
// row M=lane-16, K {8..15, 24..31}.
static __device__ __forceinline__ v16bf load_a_frag(const bf16_t* s, int rowBase,
                                                    int ld, int kBase, int lane) {
  const bf16_t* p = s + (size_t)(rowBase + (lane & 15)) * ld + kBase + ((lane >> 4) << 3);
  v16bf a;
#pragma unroll
  for (int i = 0; i < 8; ++i) { a[i] = p[i]; a[i + 8] = p[i + 16]; }
  return a;
}

// B-fragment (32x16 bf16). Column N=lane&15 is K-contiguous in LDS;
// lanes 0-15 hold K {0..7, 16..23}, lanes 16-31 hold K {8..15, 24..31}.
static __device__ __forceinline__ v16bf load_b_frag(const bf16_t* s, int colBase,
                                                    int ld, int kBase, int lane) {
  const bf16_t* p = s + (size_t)(colBase + (lane & 15)) * ld + kBase + ((lane >> 4) << 3);
  v16bf b;
#pragma unroll
  for (int i = 0; i < 8; ++i) { b[i] = p[i]; b[i + 8] = p[i + 16]; }
  return b;
}

static __device__ __forceinline__ float block_sum256(float v, float* sb) {
  __syncthreads();
#pragma unroll
  for (int off = 16; off > 0; off >>= 1) v += __shfl_xor(v, off, 32);
  int lane = threadIdx.x & 31, w = threadIdx.x >> 5;
  if (lane == 0) sb[w] = v;
  __syncthreads();
  if (w == 0) {
    float r = (lane < 8) ? sb[lane] : 0.0f;
#pragma unroll
    for (int off = 4; off > 0; off >>= 1) r += __shfl_xor(r, off, 32);
    if (lane == 0) sb[0] = r;
  }
  __syncthreads();
  return sb[0];
}

// ---------------------------------------------------------------------------
// fp32 -> bf16 elementwise convert
// ---------------------------------------------------------------------------
__global__ __launch_bounds__(256) void cvt_f32_bf16(const float* __restrict__ s,
                                                    bf16_t* __restrict__ d, int n) {
  int i = (blockIdx.x * 256 + threadIdx.x) * 4;
  if (i + 3 < n) {
    float4 v = *(const float4*)(s + i);
    d[i + 0] = (bf16_t)v.x; d[i + 1] = (bf16_t)v.y;
    d[i + 2] = (bf16_t)v.z; d[i + 3] = (bf16_t)v.w;
  }
}

// ---------------------------------------------------------------------------
// LayerNorm over a 1024-wide row -> bf16 (pre-attention q-path norm)
// ---------------------------------------------------------------------------
__global__ __launch_bounds__(256) void rowln_to_bf16(const float* __restrict__ x,
                                                     const float* __restrict__ sc,
                                                     const float* __restrict__ bi,
                                                     bf16_t* __restrict__ y) {
  __shared__ float sb[8];
  int r = blockIdx.x, t = threadIdx.x;
  const float* xr = x + (size_t)r * DMODEL;
  float lx[4]; float s = 0.0f;
#pragma unroll
  for (int i = 0; i < 4; ++i) { lx[i] = xr[t + i * 256]; s += lx[i]; }
  float mean = block_sum256(s, sb) * (1.0f / DMODEL);
  float vs = 0.0f;
#pragma unroll
  for (int i = 0; i < 4; ++i) { float d = lx[i] - mean; vs += d * d; }
  float var  = block_sum256(vs, sb) * (1.0f / DMODEL);
  float rstd = rsqrtf(var + LN_EPS);
#pragma unroll
  for (int i = 0; i < 4; ++i) {
    int c = t + i * 256;
    y[(size_t)r * DMODEL + c] = (bf16_t)((lx[i] - mean) * rstd * sc[c] + bi[c]);
  }
}

// ---------------------------------------------------------------------------
// Generic C[M,N] = A[M,K] * B[N,K]^T, bf16 in, f32 out, WMMA 16x16x32 bf16.
// Block: 256 thr = 8 waves in a 2(M)x4(N) grid; tile 128x128, BK=32.
// Staging uses global_load_async_to_lds_b128 (no VGPR round-trip).
// ---------------------------------------------------------------------------
#define BM 128
#define BN 128
#define BK 32

__global__ __launch_bounds__(256) void gemm_bf16_nt(const bf16_t* __restrict__ Amat,
                                                    const bf16_t* __restrict__ Bmat,
                                                    float* __restrict__ Cmat,
                                                    int Mdim, int Ndim, int Kdim) {
  __shared__ __attribute__((aligned(16))) bf16_t As[BM * BK];
  __shared__ __attribute__((aligned(16))) bf16_t Bs[BN * BK];
  int t = threadIdx.x, lane = t & 31, w = t >> 5;
  int wm = w >> 2, wn = w & 3;
  int m0 = blockIdx.y * BM, n0 = blockIdx.x * BN;

  v8f acc[4][2];
#pragma unroll
  for (int i = 0; i < 4; ++i)
#pragma unroll
    for (int j = 0; j < 2; ++j) acc[i][j] = vzero8();

  int lrow = t >> 1;            // 0..127
  int lkc  = (t & 1) * 16;      // 0 or 16
  unsigned aO = (unsigned)(size_t)(As + lrow * BK + lkc);
  unsigned bO = (unsigned)(size_t)(Bs + lrow * BK + lkc);
  for (int k0 = 0; k0 < Kdim; k0 += BK) {
    __syncthreads();            // previous tile fully consumed
    const bf16_t* ga = Amat + (size_t)(m0 + lrow) * Kdim + k0 + lkc;
    const bf16_t* gb = Bmat + (size_t)(n0 + lrow) * Kdim + k0 + lkc;
    async_copy_b128(aO,      ga);
    async_copy_b128(aO + 16, ga + 8);
    async_copy_b128(bO,      gb);
    async_copy_b128(bO + 16, gb + 8);
    if (k0 + BK < Kdim) {       // global_prefetch_b8 of next k-tile
      __builtin_prefetch(ga + BK, 0, 1);
      __builtin_prefetch(gb + BK, 0, 1);
    }
    wait_async0();
    __syncthreads();
#pragma unroll
    for (int mf = 0; mf < 4; ++mf) {
      v16bf a = load_a_frag(As, wm * 64 + mf * 16, BK, 0, lane);
#pragma unroll
      for (int nf = 0; nf < 2; ++nf) {
        v16bf b = load_b_frag(Bs, wn * 32 + nf * 16, BK, 0, lane);
        acc[mf][nf] = __builtin_amdgcn_wmma_f32_16x16x32_bf16(
            false, a, false, b, (short)0, acc[mf][nf], false, false);
      }
    }
  }
  // C layout (ISA 7.12.2): VGPR i -> row i (+8 for lanes 16-31), col = lane&15
  int hi8 = (lane >> 4) * 8, cl = lane & 15;
#pragma unroll
  for (int mf = 0; mf < 4; ++mf)
#pragma unroll
    for (int nf = 0; nf < 2; ++nf)
#pragma unroll
      for (int i = 0; i < 8; ++i) {
        int r = m0 + wm * 64 + mf * 16 + i + hi8;
        int c = n0 + wn * 32 + nf * 16 + cl;
        Cmat[(size_t)r * Ndim + c] = acc[mf][nf][i];
      }
}

// ---------------------------------------------------------------------------
// Per-head repack: proj fp32 [B,N,A] -> bf16 [B,H,N,64]; optional 64-wide
// LayerNorm * SIG_SCALE (q/k) or plain convert (v). One wave per head-row.
// ---------------------------------------------------------------------------
__global__ __launch_bounds__(256) void qk_repack(const float* __restrict__ proj,
                                                 bf16_t* __restrict__ outp,
                                                 const float* __restrict__ sc,
                                                 const float* __restrict__ bi,
                                                 int doNorm) {
  int slice = blockIdx.x * 8 + (threadIdx.x >> 5);
  int lane  = threadIdx.x & 31;
  int h  = slice & 15;
  int bn = slice >> 4;                         // b*NQ + n
  const float* src = proj + (size_t)bn * AMODEL + h * HEAD_DIM;
  float x0 = src[lane * 2], x1 = src[lane * 2 + 1];
  float y0, y1;
  if (doNorm) {
    float s = x0 + x1;
#pragma unroll
    for (int off = 16; off > 0; off >>= 1) s += __shfl_xor(s, off, 32);
    float mean = s * (1.0f / HEAD_DIM);
    float d0 = x0 - mean, d1 = x1 - mean;
    float v = d0 * d0 + d1 * d1;
#pragma unroll
    for (int off = 16; off > 0; off >>= 1) v += __shfl_xor(v, off, 32);
    float rstd = rsqrtf(v * (1.0f / HEAD_DIM) + LN_EPS);
    y0 = (d0 * rstd * sc[lane * 2]     + bi[lane * 2])     * SIG_SCALE;
    y1 = (d1 * rstd * sc[lane * 2 + 1] + bi[lane * 2 + 1]) * SIG_SCALE;
  } else {
    y0 = x0; y1 = x1;
  }
  int b = bn >> 11, n = bn & (NQ - 1);
  size_t o = (((size_t)(b * HEADS + h)) * NQ + n) * HEAD_DIM + lane * 2;
  outp[o]     = (bf16_t)y0;
  outp[o + 1] = (bf16_t)y1;
}

// ---------------------------------------------------------------------------
// Fused sigmoid attention: per (b,h,128-row q block), stream 64-wide K/V
// tiles. No softmax -> single-pass accumulate. 8 waves, each owns 16 q rows.
// Q tile via TDM (tensor_load_to_lds), K tile via async-LDS loads.
// ---------------------------------------------------------------------------
__global__ __launch_bounds__(256) void attn_sigmoid(const bf16_t* __restrict__ Qh,
                                                    const bf16_t* __restrict__ Kh,
                                                    const bf16_t* __restrict__ Vh,
                                                    bf16_t* __restrict__ Ctx) {
  __shared__ __attribute__((aligned(16))) bf16_t Qs[128 * HEAD_DIM];   // 16 KB
  __shared__ __attribute__((aligned(16))) bf16_t Ks[64 * HEAD_DIM];    //  8 KB
  __shared__ __attribute__((aligned(16))) bf16_t Vst[HEAD_DIM * 64];   //  8 KB ([d][k])
  __shared__ __attribute__((aligned(16))) bf16_t Ss[128 * 64];         // 16 KB (P staging)

  int t = threadIdx.x, lane = t & 31, w = t >> 5;
  const int nqt = NQ / 128;
  int qt = blockIdx.x % nqt;
  int h  = (blockIdx.x / nqt) % HEADS;
  int b  = blockIdx.x / (nqt * HEADS);
  size_t headBase = (size_t)(b * HEADS + h) * NQ * HEAD_DIM;

  // ---- load Q block (128x64 bf16 = 16 KB contiguous) ----
  const bf16_t* qg = Qh + headBase + (size_t)qt * 128 * HEAD_DIM;
#if __has_builtin(__builtin_amdgcn_tensor_load_to_lds) && __has_builtin(__builtin_amdgcn_s_wait_tensorcnt)
  if (w == 0) {  // one TDM descriptor per workgroup; EXEC-independent DMA
    typedef unsigned int v4u __attribute__((ext_vector_type(4)));
    typedef int v4i __attribute__((ext_vector_type(4)));
    typedef int v8i __attribute__((ext_vector_type(8)));
    unsigned long long gaddr = (unsigned long long)(size_t)qg;
    v4u g0;
    g0[0] = 1u;                                   // count=1, user descriptor
    g0[1] = (unsigned)(size_t)(void*)Qs;          // lds_addr (bytes)
    g0[2] = (unsigned)gaddr;                      // global_addr[31:0]
    g0[3] = (unsigned)((gaddr >> 32) & 0x1FFFFFFu) | (2u << 30);  // addr[56:32] | type=2
    v8i g1;
    g1[0] = (int)(1u << 16);                      // data_size=1 (2 bytes/elem)
    g1[1] = (int)(8192u << 16);                   // tensor_dim0 = 8192 (low 16)
    g1[2] = (int)(1u << 16);                      // tensor_dim0 hi=0 | tensor_dim1=1
    g1[3] = (int)(8192u << 16);                   // tensor_dim1 hi=0 | tile_dim0=8192
    g1[4] = 0;                                    // tile_dim1=0, tile_dim2=0
    g1[5] = 8192;                                 // tensor_dim0_stride low32
    g1[6] = 0; g1[7] = 0;
    v4i gz4 = {0, 0, 0, 0};
    v8i gz8 = {0, 0, 0, 0, 0, 0, 0, 0};
    __builtin_amdgcn_tensor_load_to_lds(g0, g1, gz4, gz4, gz8, 0);
    __builtin_amdgcn_s_wait_tensorcnt(0);
  }
#else
#pragma unroll
  for (int j = 0; j < 4; ++j) {
    int c = t * 4 + j;
    async_copy_b128((unsigned)(size_t)(Qs + c * 8), qg + c * 8);
  }
  wait_async0();
#endif

  v8f acc[4];
#pragma unroll
  for (int i = 0; i < 4; ++i) acc[i] = vzero8();

  for (int kt = 0; kt < NKK / 64; ++kt) {
    __syncthreads();
    // K tile via async memory->LDS (64x64 contiguous)
    const bf16_t* kg = Kh + headBase + (size_t)kt * 64 * HEAD_DIM;
    async_copy_b128((unsigned)(size_t)(Ks + (t * 2) * 8),     kg + (t * 2) * 8);
    async_copy_b128((unsigned)(size_t)(Ks + (t * 2 + 1) * 8), kg + (t * 2 + 1) * 8);
    // V tile transposed into [d][k] so B-fragments are K-contiguous
    const bf16_t* vg = Vh + headBase + (size_t)kt * 64 * HEAD_DIM;
    {
      int vn  = t >> 2;
      int vd0 = (t & 3) << 4;
      bf16_t tmp[16];
      *(uint4*)(tmp)     = *(const uint4*)(vg + vn * HEAD_DIM + vd0);
      *(uint4*)(tmp + 8) = *(const uint4*)(vg + vn * HEAD_DIM + vd0 + 8);
#pragma unroll
      for (int j = 0; j < 16; ++j) Vst[(vd0 + j) * 64 + vn] = tmp[j];
    }
    wait_async0();
    __syncthreads();

    // S = Q * K^T  (contraction over d = 64 -> two chained WMMAs)
    int hi8 = (lane >> 4) * 8, cl = lane & 15;
#pragma unroll
    for (int nf = 0; nf < 4; ++nf) {
      v8f s = vzero8();
#pragma unroll
      for (int kk = 0; kk < HEAD_DIM; kk += 32) {
        v16bf a = load_a_frag(Qs, w * 16, HEAD_DIM, kk, lane);
        v16bf bb = load_b_frag(Ks, nf * 16, HEAD_DIM, kk, lane);
        s = __builtin_amdgcn_wmma_f32_16x16x32_bf16(false, a, false, bb,
                                                    (short)0, s, false, false);
      }
      // sigmoid(score - 3) via 0.5*(1+tanh(0.5 x)); stage P bf16 in LDS
#pragma unroll
      for (int i = 0; i < 8; ++i) {
        float x = s[i] - 3.0f;
        float p = 0.5f * (1.0f + tanhf(0.5f * x));
        Ss[(w * 16 + i + hi8) * 64 + nf * 16 + cl] = (bf16_t)p;
      }
    }
    __syncthreads();

    // ctx += P * V  (contraction over k-tile = 64 -> two chained WMMAs)
#pragma unroll
    for (int kc = 0; kc < 2; ++kc) {
      v16bf a = load_a_frag(Ss, w * 16, 64, kc * 32, lane);
#pragma unroll
      for (int cn = 0; cn < 4; ++cn) {
        v16bf bb = load_b_frag(Vst, cn * 16, 64, kc * 32, lane);
        acc[cn] = __builtin_amdgcn_wmma_f32_16x16x32_bf16(false, a, false, bb,
                                                          (short)0, acc[cn], false, false);
      }
    }
  }

  // write ctx in [B, Nq, H*64] so the output projection is a plain GEMM
  int hi8 = (lane >> 4) * 8, cl = lane & 15;
#pragma unroll
  for (int cn = 0; cn < 4; ++cn)
#pragma unroll
    for (int i = 0; i < 8; ++i) {
      int n = qt * 128 + w * 16 + i + hi8;
      int d = cn * 16 + cl;
      Ctx[((size_t)(b * NQ + n)) * AMODEL + h * HEAD_DIM + d] = (bf16_t)acc[cn][i];
    }
}

// ---------------------------------------------------------------------------
// out = LayerNorm(residual + gamma * proj) -> fp32
// ---------------------------------------------------------------------------
__global__ __launch_bounds__(256) void final_ln(const float* __restrict__ resid,
                                                const float* __restrict__ proj,
                                                const float* __restrict__ gamma,
                                                const float* __restrict__ sc,
                                                const float* __restrict__ bi,
                                                float* __restrict__ out) {
  __shared__ float sb[8];
  int r = blockIdx.x, t = threadIdx.x;
  const float* rr = resid + (size_t)r * DMODEL;
  const float* pr = proj  + (size_t)r * DMODEL;
  float lx[4]; float s = 0.0f;
#pragma unroll
  for (int i = 0; i < 4; ++i) {
    int c = t + i * 256;
    lx[i] = rr[c] + gamma[c] * pr[c];
    s += lx[i];
  }
  float mean = block_sum256(s, sb) * (1.0f / DMODEL);
  float vs = 0.0f;
#pragma unroll
  for (int i = 0; i < 4; ++i) { float d = lx[i] - mean; vs += d * d; }
  float var  = block_sum256(vs, sb) * (1.0f / DMODEL);
  float rstd = rsqrtf(var + LN_EPS);
#pragma unroll
  for (int i = 0; i < 4; ++i) {
    int c = t + i * 256;
    out[(size_t)r * DMODEL + c] = (lx[i] - mean) * rstd * sc[c] + bi[c];
  }
}

// ---------------------------------------------------------------------------
// launch
// ---------------------------------------------------------------------------
extern "C" void kernel_launch(void* const* d_in, const int* in_sizes, int n_in,
                              void* d_out, int out_size, void* d_ws, size_t ws_size,
                              hipStream_t stream) {
  const float* qf    = (const float*)d_in[0];
  const float* kf    = (const float*)d_in[1];
  const float* Wq    = (const float*)d_in[2];
  const float* Wk    = (const float*)d_in[3];
  const float* Wv    = (const float*)d_in[4];
  const float* Wo    = (const float*)d_in[5];
  const float* qns   = (const float*)d_in[6];
  const float* qnb   = (const float*)d_in[7];
  const float* kns   = (const float*)d_in[8];
  const float* knb   = (const float*)d_in[9];
  const float* lqs   = (const float*)d_in[10];
  const float* lqb   = (const float*)d_in[11];
  const float* los   = (const float*)d_in[12];
  const float* lob   = (const float*)d_in[13];
  const float* gamma = (const float*)d_in[14];
  float* outp = (float*)d_out;

  char* ws = (char*)d_ws;
  const size_t MB = 1024 * 1024;
  bf16_t* xq   = (bf16_t*)(ws + 0 * MB);    // 8 MB  [4096,1024]
  bf16_t* kfb  = (bf16_t*)(ws + 8 * MB);    // 8 MB
  bf16_t* wqb  = (bf16_t*)(ws + 16 * MB);   // 2 MB
  bf16_t* wkb  = (bf16_t*)(ws + 18 * MB);   // 2 MB
  bf16_t* wvb  = (bf16_t*)(ws + 20 * MB);   // 2 MB
  bf16_t* wob  = (bf16_t*)(ws + 22 * MB);   // 2 MB
  float*  proj = (float*) (ws + 24 * MB);   // 16 MB (reused q/k/v/out proj)
  bf16_t* qh   = (bf16_t*)(ws + 40 * MB);   // 8 MB  [B,H,N,64]
  bf16_t* kh   = (bf16_t*)(ws + 48 * MB);   // 8 MB
  bf16_t* vh   = (bf16_t*)(ws + 56 * MB);   // 8 MB
  bf16_t* ctx  = (bf16_t*)(ws + 64 * MB);   // 8 MB

  // bf16 staging
  cvt_f32_bf16<<<4096, 256, 0, stream>>>(kf, kfb, MROWS * DMODEL);
  cvt_f32_bf16<<<1024, 256, 0, stream>>>(Wq, wqb, AMODEL * DMODEL);
  cvt_f32_bf16<<<1024, 256, 0, stream>>>(Wk, wkb, AMODEL * DMODEL);
  cvt_f32_bf16<<<1024, 256, 0, stream>>>(Wv, wvb, AMODEL * DMODEL);
  cvt_f32_bf16<<<1024, 256, 0, stream>>>(Wo, wob, DMODEL * AMODEL);
  rowln_to_bf16<<<MROWS, 256, 0, stream>>>(qf, lqs, lqb, xq);

  dim3 ggrid(AMODEL / BN, MROWS / BM);
  // Q projection + qk-norm repack
  gemm_bf16_nt<<<ggrid, 256, 0, stream>>>(xq, wqb, proj, MROWS, AMODEL, DMODEL);
  qk_repack<<<(MROWS * HEADS) / 8, 256, 0, stream>>>(proj, qh, qns, qnb, 1);
  // K projection + kn-norm repack
  gemm_bf16_nt<<<ggrid, 256, 0, stream>>>(kfb, wkb, proj, MROWS, AMODEL, DMODEL);
  qk_repack<<<(MROWS * HEADS) / 8, 256, 0, stream>>>(proj, kh, kns, knb, 1);
  // V projection + plain repack
  gemm_bf16_nt<<<ggrid, 256, 0, stream>>>(kfb, wvb, proj, MROWS, AMODEL, DMODEL);
  qk_repack<<<(MROWS * HEADS) / 8, 256, 0, stream>>>(proj, vh, qns, qnb, 0);

  // fused sigmoid attention
  attn_sigmoid<<<BATCH * HEADS * (NQ / 128), 256, 0, stream>>>(qh, kh, vh, ctx);

  // output projection + residual/gamma/post-LN
  gemm_bf16_nt<<<ggrid, 256, 0, stream>>>(ctx, wob, proj, MROWS, DMODEL, AMODEL);
  final_ln<<<MROWS, 256, 0, stream>>>(qf, proj, gamma, los, lob, outp);
}
